// ContextQuestionAttention_89936615178704
// MI455X (gfx1250) — compile-verified, matrix operand order checked by reading, not run on hardware
//
#include <hip/hip_runtime.h>

#define NEGV (-1000000000.0f)

typedef __attribute__((ext_vector_type(16))) __bf16 v16bf;
typedef __attribute__((ext_vector_type(8)))  __bf16 bfx8;
typedef __attribute__((ext_vector_type(8)))  float  v8f;
typedef __attribute__((ext_vector_type(4)))  int    v4i;

#define AS_GLOBAL __attribute__((address_space(1)))
#define AS_LOCAL  __attribute__((address_space(3)))

#if __has_builtin(__builtin_amdgcn_global_load_async_to_lds_b128)
#define HAVE_ASYNC_LDS 1
#else
#define HAVE_ASYNC_LDS 0
#endif

__device__ __forceinline__ v8f wmma_bf16(v16bf a, v16bf b, v8f c) {
    // D(16x16 f32) = A(16x32 bf16) x B(32x16 bf16) + C
    return __builtin_amdgcn_wmma_f32_16x16x32_bf16(
        /*neg_a=*/false, a, /*neg_b=*/false, b,
        /*c_mod=*/(short)0, c, /*reuse_a=*/false, /*reuse_b=*/false);
}

// ---------------------------------------------------------------------------
// Kernel 1: per (64 C-rows, batch) block; 4 waves, one 16-row stripe each.
//   S = (C*w3)@Q^T + (C@w1) + s_q -> mask -> softmax -> A = P@Q
//   writes out[:,0:512]=C, [512:1024]=A, [1024:1536]=C*A (all NT stores),
//   maxS -> ws.  Q staged to LDS (bf16, dual layout) via async-to-LDS chunks.
// LDS map (dynamic, 316672 B total):
//   Qs[128*512]bf16 @0 | Qt[512*128]bf16 @131072 | Ps[4][16*128]bf16 @262144
//   stage[16*512]f32 @278528 | sq[128] @311296 | w3s[512] @311808
//   w1s[512] @313856 | scs[64] @315904 | qms[128] @316160
// ---------------------------------------------------------------------------
__global__ __launch_bounds__(128) void attn_kernel(
    const float* __restrict__ Cg, const float* __restrict__ Qg,
    const int* __restrict__ qmask, const float* __restrict__ w,
    float* __restrict__ out, float* __restrict__ maxs)
{
    constexpr int CL = 1024, QL = 128, H = 512;
    extern __shared__ char smem[];
    __bf16* Qs    = (__bf16*)smem;
    __bf16* Qt    = (__bf16*)(smem + 131072);
    __bf16* Ps    = (__bf16*)(smem + 262144);
    float*  stage = (float*)(smem + 278528);
    float*  sq    = (float*)(smem + 311296);
    float*  w3s   = (float*)(smem + 311808);
    float*  w1s   = (float*)(smem + 313856);
    float*  scs   = (float*)(smem + 315904);
    int*    qms   = (int*)(smem + 316160);

    const int b    = blockIdx.y;
    const int t    = threadIdx.x;                  // 128 threads = 4 waves
    const int wave = t >> 5;
    const int lane = t & 31;
    const int half = lane >> 4;
    const int lgrp = lane & 15;
    const int m0w  = blockIdx.x * 64 + wave * 16;  // this wave's C-row base

    // ---- stage w1, w3 ----
    for (int i = t; i < H; i += 128) { w1s[i] = w[i]; w3s[i] = w[2 * H + i]; }

    // ---- s_q = Q@w2 and q_mask (one Q row per thread) ----
    {
        qms[t] = qmask[b * QL + t];
        const float* qrow = Qg + ((size_t)b * QL + t) * H;
        float s = 0.f;
        for (int k = 0; k < H; k += 4) {
            float4 q4 = *(const float4*)(qrow + k);
            float4 w4 = *(const float4*)(w + H + k);
            s += q4.x * w4.x + q4.y * w4.y + q4.z * w4.z + q4.w * w4.w;
        }
        sq[t] = s;
    }

    // ---- stage Q as bf16 in both layouts, via async-to-LDS f32 chunks ----
    for (int chunk = 0; chunk < 8; chunk++) {       // 16 Q rows per chunk
        const float* src = Qg + (size_t)b * QL * H + chunk * 16 * H;
#if HAVE_ASYNC_LDS
        for (int q4 = t; q4 < 16 * H / 4; q4 += 128) {
            __builtin_amdgcn_global_load_async_to_lds_b128(
                (AS_GLOBAL v4i*)(src + q4 * 4),
                (AS_LOCAL v4i*)(stage + q4 * 4), 0, 0);
        }
        asm volatile("s_wait_asynccnt 0x0" ::: "memory");
#else
        for (int e = t; e < 16 * H; e += 128) stage[e] = src[e];
#endif
        __syncthreads();
        for (int e = t; e < 16 * H; e += 128) {
            int jr = e >> 9, h = e & (H - 1);
            __bf16 v = (__bf16)stage[e];
            int j = chunk * 16 + jr;
            Qs[j * H + h] = v;
            Qt[h * QL + j] = v;
        }
        __syncthreads();
    }

    // ---- GEMM1: S stripe 16x128 in 8 accumulators; fuse s_c = C@w1 ----
    v8f accS[8];
#pragma unroll
    for (int nt = 0; nt < 8; nt++) accS[nt] = (v8f){0,0,0,0,0,0,0,0};
    float scAcc = 0.f;                              // partial C@w1, row=lgrp

    const size_t cRowOff = ((size_t)b * CL + m0w + lgrp) * H;
    for (int kb = 0; kb < H; kb += 32) {
        const int k0 = kb + half * 8;               // A layout: half picks K+-8
        __builtin_prefetch(Cg + cRowOff + k0 + 128, 0, 1);
        v16bf afrag;
        {
            const float4* c4 = (const float4*)(Cg + cRowOff + k0);
            float av[16];
            *(float4*)&av[0]  = c4[0];  *(float4*)&av[4]  = c4[1];   // K k0..+7
            *(float4*)&av[8]  = c4[4];  *(float4*)&av[12] = c4[5];   // K k0+16..+23
#pragma unroll
            for (int i = 0; i < 16; i++) {
                int kk = k0 + (i & 7) + ((i >> 3) << 4);
                scAcc += av[i] * w1s[kk];
                afrag[i] = (__bf16)(av[i] * w3s[kk]);
            }
        }
#pragma unroll
        for (int nt = 0; nt < 8; nt++) {
            const __bf16* qp = Qs + (nt * 16 + lgrp) * H + k0;  // B[k][n]=Q[n][k]
            bfx8 b0 = *(const bfx8*)qp;
            bfx8 b1 = *(const bfx8*)(qp + 16);
            v16bf bfrag;
#pragma unroll
            for (int i = 0; i < 8; i++) { bfrag[i] = b0[i]; bfrag[i + 8] = b1[i]; }
            accS[nt] = wmma_bf16(afrag, bfrag, accS[nt]);
        }
    }

    // ---- publish s_c (row = lgrp holds full sum after xor-16) ----
    scAcc += __shfl_xor(scAcc, 16, 32);
    if (half == 0) scs[wave * 16 + lgrp] = scAcc;
    __syncthreads();

    // ---- bias + mask; D layout: row = i + half*8, col = nt*16+lgrp ----
#pragma unroll
    for (int nt = 0; nt < 8; nt++) {
        const int col = nt * 16 + lgrp;
        const float sqv = sq[col];
        const int mk = qms[col];
#pragma unroll
        for (int i = 0; i < 8; i++) {
            float v = accS[nt][i] + scs[wave * 16 + i + half * 8] + sqv;
            accS[nt][i] = mk ? v : NEGV;
        }
    }
    float rmax[8];
#pragma unroll
    for (int i = 0; i < 8; i++) {
        float m = accS[0][i];
#pragma unroll
        for (int nt = 1; nt < 8; nt++) m = fmaxf(m, accS[nt][i]);
        m = fmaxf(m, __shfl_xor(m, 1, 32));
        m = fmaxf(m, __shfl_xor(m, 2, 32));
        m = fmaxf(m, __shfl_xor(m, 4, 32));
        m = fmaxf(m, __shfl_xor(m, 8, 32));
        rmax[i] = m;
    }
    if (lgrp == 0) {
#pragma unroll
        for (int i = 0; i < 8; i++)
            maxs[(size_t)b * CL + m0w + i + half * 8] = rmax[i];
    }
    float inv[8];
#pragma unroll
    for (int i = 0; i < 8; i++) {
        float s = 0.f;
#pragma unroll
        for (int nt = 0; nt < 8; nt++) {
            float p = __expf(accS[nt][i] - rmax[i]);
            accS[nt][i] = p;
            s += p;
        }
        s += __shfl_xor(s, 1, 32);
        s += __shfl_xor(s, 2, 32);
        s += __shfl_xor(s, 4, 32);
        s += __shfl_xor(s, 8, 32);
        inv[i] = 1.0f / s;
    }

    // ---- spill P (unnormalized) to LDS bf16, per-wave region ----
    __bf16* Pw = Ps + wave * (16 * QL);
#pragma unroll
    for (int nt = 0; nt < 8; nt++)
#pragma unroll
        for (int i = 0; i < 8; i++)
            Pw[(i + half * 8) * QL + nt * 16 + lgrp] = (__bf16)accS[nt][i];
    __syncthreads();

    // ---- GEMM2: A = P @ Q; N=512 (32 tiles), K=128 (4 chunks) ----
    v16bf pa[4];
#pragma unroll
    for (int kc = 0; kc < 4; kc++) {
        const int k0 = kc * 32 + half * 8;
        bfx8 p0 = *(const bfx8*)(Pw + lgrp * QL + k0);
        bfx8 p1 = *(const bfx8*)(Pw + lgrp * QL + k0 + 16);
#pragma unroll
        for (int i = 0; i < 8; i++) { pa[kc][i] = p0[i]; pa[kc][i + 8] = p1[i]; }
    }
    for (int nt = 0; nt < 32; nt++) {
        v8f acc = (v8f){0,0,0,0,0,0,0,0};
        const int h = nt * 16 + lgrp;
#pragma unroll
        for (int kc = 0; kc < 4; kc++) {
            const int k0 = kc * 32 + half * 8;
            bfx8 q0 = *(const bfx8*)(Qt + h * QL + k0);   // B[k][n]=Q[k][n]=Qt[n][k]
            bfx8 q1 = *(const bfx8*)(Qt + h * QL + k0 + 16);
            v16bf bfrag;
#pragma unroll
            for (int i = 0; i < 8; i++) { bfrag[i] = q0[i]; bfrag[i + 8] = q1[i]; }
            acc = wmma_bf16(pa[kc], bfrag, acc);
        }
#pragma unroll
        for (int i = 0; i < 8; i++) {
            const int r = m0w + i + half * 8;
            const float aval = acc[i] * inv[i];
            const size_t obase = ((size_t)b * CL + r) * 2048;
            const float cv = Cg[((size_t)b * CL + r) * H + h];   // L2-hot
            __builtin_nontemporal_store(cv,          &out[obase + h]);
            __builtin_nontemporal_store(aval,        &out[obase + 512 + h]);
            __builtin_nontemporal_store(cv * aval,   &out[obase + 1024 + h]);
        }
    }
}

// ---------------------------------------------------------------------------
// Kernel 2: B_att = softmax(maxS) over C_len; B_vec = B_att . C
// grid (4, B): block g owns h in [g*128, g*128+128); 2 threads per h split i.
// ---------------------------------------------------------------------------
__global__ __launch_bounds__(256) void bvec_kernel(
    const float* __restrict__ Cg, const float* __restrict__ maxs,
    float* __restrict__ bvec)
{
    constexpr int CL = 1024, H = 512;
    __shared__ float p[CL];
    __shared__ float red[256];
    const int b = blockIdx.y, g = blockIdx.x, t = threadIdx.x;

    float lm = -3.0e38f;
    for (int i = t; i < CL; i += 256) { float v = maxs[(size_t)b * CL + i]; p[i] = v; lm = fmaxf(lm, v); }
    red[t] = lm; __syncthreads();
    for (int s = 128; s > 0; s >>= 1) { if (t < s) red[t] = fmaxf(red[t], red[t + s]); __syncthreads(); }
    const float mx = red[0]; __syncthreads();

    float ls = 0.f;
    for (int i = t; i < CL; i += 256) { float e = __expf(p[i] - mx); p[i] = e; ls += e; }
    red[t] = ls; __syncthreads();
    for (int s = 128; s > 0; s >>= 1) { if (t < s) red[t] += red[t + s]; __syncthreads(); }
    const float inv = 1.0f / red[0];
    __syncthreads();

    const int h  = g * 128 + (t & 127);
    const int i0 = (t >> 7) * 512;
    float acc = 0.f;
#pragma unroll 4
    for (int i = 0; i < 512; i++)
        acc += p[i0 + i] * Cg[((size_t)b * CL + i0 + i) * H + h];
    red[t] = acc; __syncthreads();
    if (t < 128) bvec[(size_t)b * H + h] = (red[t] + red[t + 128]) * inv;
}

// ---------------------------------------------------------------------------
// Kernel 3: out[:,:,1536:2048] = C * B_vec  (C is L2-resident; NT stores)
// ---------------------------------------------------------------------------
__global__ __launch_bounds__(256) void cbvec_kernel(
    const float* __restrict__ Cg, const float* __restrict__ bvec,
    float* __restrict__ out)
{
    const size_t idx = (size_t)blockIdx.x * 256 + threadIdx.x; // B*CL*H
    const int h = (int)(idx & 511);
    const size_t bi = idx >> 9;           // b*CL + i
    const int b = (int)(bi >> 10);
    const float c = Cg[idx];
    const float bv = bvec[(size_t)b * 512 + h];
    __builtin_nontemporal_store(c * bv, &out[bi * 2048 + 1536 + h]);
}

extern "C" void kernel_launch(void* const* d_in, const int* in_sizes, int n_in,
                              void* d_out, int out_size, void* d_ws, size_t ws_size,
                              hipStream_t stream) {
    const float* C     = (const float*)d_in[0];   // (32,1024,512)
    const float* Q     = (const float*)d_in[1];   // (32,128,512)
    const int*   qmask = (const int*)d_in[2];     // (32,128)
    const float* w     = (const float*)d_in[3];   // (1536,)
    float* out = (float*)d_out;                   // (32,1024,2048)
    float* ws  = (float*)d_ws;
    float* maxs = ws;                // 32*1024 floats
    float* bvec = ws + 32 * 1024;    // 32*512 floats

    const size_t smem1 = 316672;     // see attn_kernel partition map
    dim3 g1(16, 32);
    attn_kernel<<<g1, 128, smem1, stream>>>(C, Q, qmask, w, out, maxs);
    dim3 g2(4, 32);
    bvec_kernel<<<g2, 256, 0, stream>>>(C, maxs, bvec);
    cbvec_kernel<<<65536, 256, 0, stream>>>(C, bvec, out);
}
